// SuperpositionDenoisingLayer_1778116461105
// MI455X (gfx1250) — matrix operand order repaired and checked by reading, not compiled, not run on hardware
//
#include <hip/hip_runtime.h>
#include <hip/hip_bf16.h>

#define E_ 200000
#define N_ 25000

typedef __attribute__((ext_vector_type(16))) __bf16 v16bf;
typedef __attribute__((ext_vector_type(8)))  float  v8f;

#define DEV __device__ __forceinline__

DEV unsigned short bfbits(float x) {
  unsigned u = __float_as_uint(x);
  unsigned r = u + 0x7FFFu + ((u >> 16) & 1u);   // round-to-nearest-even
  return (unsigned short)(r >> 16);
}
DEV __bf16 f2bf(float x) {
  unsigned short h = bfbits(x);
  __bf16 b; __builtin_memcpy(&b, &h, 2); return b;
}

DEV v8f vzero() { v8f z;
#pragma unroll
  for (int i = 0; i < 8; ++i) z[i] = 0.f;
  return z; }

DEV v8f wmma_bf16(v16bf a, v16bf b, v8f c) {
  return __builtin_amdgcn_wmma_f32_16x16x32_bf16(false, a, false, b, (short)0, c, false, false);
}

// ---- fragment-order layouts: each lane's 16 bf16 values are contiguous (32B) ----
// A-tile in LDS:  [kt][lane][16]   (512 bf16 per k-tile)
// B-weights glob: [kt][nt][lane][16]
// (row,col%32=cc) -> flane = row + 16*((cc>>3)&1), e = (cc&7) + 8*(cc>>4)

DEV v16bf frag_a(const __bf16* tile, int kt, int lane) {
  return *(const v16bf*)(tile + kt * 512 + lane * 16);
}
DEV v16bf frag_b(const __bf16* Wf, int kt, int nt, int nNT, int lane) {
  return *(const v16bf*)(Wf + ((long)(kt * nNT + nt) * 512) + lane * 16);
}

DEV void tile_store1(__bf16* tile, int row, int col, float v) {
  int kt = col >> 5, cc = col & 31;
  int fl = row + (((cc >> 3) & 1) << 4);
  int e = (cc & 7) + ((cc >> 4) << 3);
  tile[kt * 512 + fl * 16 + e] = f2bf(v);
}

// fill a 16xK tile (K multiple of 32) in fragment order; getv(row,col)->float
template <typename F>
DEV void fill_tile(__bf16* tile, int K, int lane, F getv) {
  int quads = (16 * K) >> 2;
  for (int g = lane; g < quads; g += 32) {
    int idx = g << 2;
    int row = idx / K, col = idx % K;           // col % 4 == 0
    float x0 = getv(row, col + 0);
    float x1 = getv(row, col + 1);
    float x2 = getv(row, col + 2);
    float x3 = getv(row, col + 3);
    int kt = col >> 5, cc = col & 31;
    int fl = row + (((cc >> 3) & 1) << 4);
    int e = (cc & 7) + ((cc >> 4) << 3);        // e % 4 == 0
    unsigned lo = ((unsigned)bfbits(x1) << 16) | bfbits(x0);
    unsigned hi = ((unsigned)bfbits(x3) << 16) | bfbits(x2);
    unsigned* q = (unsigned*)(tile + kt * 512 + fl * 16 + e);
    q[0] = lo; q[1] = hi;
  }
}

DEV float wave_sum(float v) {
#pragma unroll
  for (int o = 16; o > 0; o >>= 1) v += __shfl_xor(v, o, 32);
  return v;
}

DEV float fast_rcp(float x) { return __builtin_amdgcn_rcpf(x); }
DEV float silu(float x) { return x * fast_rcp(1.f + __expf(-x)); }

DEV void atomicMaxF(float* a, float v) {
  if (v >= 0.f) atomicMax((int*)a, __float_as_int(v));
  else          atomicMin((unsigned int*)a, __float_as_uint(v));
}

// ---------------- weight convert: f32 KxNsrc -> bf16 fragment-order [Kp/32][Np/16][32][16] ----------------
__global__ void k_cvt(const float* s, __bf16* d, int K, int Nsrc, int Kp, int Np) {
  int i = blockIdx.x * blockDim.x + threadIdx.x;
  if (i >= Kp * Np) return;
  int frag = i >> 9, r = i & 511;
  int lane = r >> 4, e = r & 15;
  int nNT = Np >> 4;
  int kt = frag / nNT, nt = frag % nNT;
  int k = kt * 32 + ((e >> 3) << 4) + ((lane >> 4) << 3) + (e & 7);
  int n = nt * 16 + (lane & 15);
  float v = (k < K && n < Nsrc) ? s[(long)k * Nsrc + n] : 0.f;
  d[i] = f2bf(v);
}

__global__ void k_fill(float* p, float v, long n) {
  long i = (long)blockIdx.x * blockDim.x + threadIdx.x;
  if (i < n) p[i] = v;
}

__global__ void k_padvec(const float* s, float* d, int n, int np) {
  int i = blockIdx.x * blockDim.x + threadIdx.x;
  if (i < np) d[i] = (i < n) ? s[i] : 0.f;
}

// ---------------- edge linear: silu(ef @ W + b), K=128, Np in {96,32}; bias length >= Np ----------------
__global__ __launch_bounds__(256) void k_edge_lin(const float* ef, const __bf16* W,
                                                  const float* bias, float* outp,
                                                  int Np, int ldo) {
  __shared__ __bf16 tile[8][16 * 128] __attribute__((aligned(32)));
  int w = threadIdx.x >> 5, lane = threadIdx.x & 31;
  long e0 = (long)(blockIdx.x * 8 + w) * 16;
  bool full = (e0 + 16 <= E_);
  if (full) {
    __builtin_prefetch(ef + (e0 + 16) * 128, 0, 1);
    fill_tile(tile[w], 128, lane, [&](int row, int col) -> float {
      return ef[(e0 + row) * 128 + col];
    });
  } else {
    fill_tile(tile[w], 128, lane, [&](int row, int col) -> float {
      long e = e0 + row;
      return e < E_ ? ef[e * 128 + col] : 0.f;
    });
  }
  __syncthreads();
  int nNT = Np >> 4;
  for (int nt = 0; nt < nNT; ++nt) {
    v8f acc = vzero();
#pragma unroll
    for (int kt = 0; kt < 4; ++kt)
      acc = wmma_bf16(frag_a(tile[w], kt, lane), frag_b(W, kt, nt, nNT, lane), acc);
    int col = nt * 16 + (lane & 15), hi = (lane >> 4) << 3;
    float b = bias[col];
    if (full) {
#pragma unroll
      for (int r = 0; r < 8; ++r)
        outp[(e0 + r + hi) * ldo + col] = silu(acc[r] + b);
    } else {
#pragma unroll
      for (int r = 0; r < 8; ++r) {
        long e = e0 + r + hi;
        if (e < E_) outp[e * ldo + col] = silu(acc[r] + b);
      }
    }
  }
}

// ---------------- atom->node messages: (atom[src]*rad) @ W_a2n, 4 edges/wave ----------------
__global__ __launch_bounds__(256) void k_a2n(const float* atom, const float* rad,
                                             const int* src, const int* dst,
                                             const __bf16* W, float* a2n) {
  __shared__ __bf16 tile[8][16 * 96] __attribute__((aligned(32)));
  int w = threadIdx.x >> 5, lane = threadIdx.x & 31;
  long e0 = (long)(blockIdx.x * 8 + w) * 4;
  bool full = (e0 + 4 <= E_);
  if (full) {
    fill_tile(tile[w], 96, lane, [&](int row, int col) -> float {
      long e = e0 + (row >> 2); int m = row & 3;
      if (col < 91) {
        long s = src[e];
        return atom[(s * 4 + m) * 91 + col] * rad[e * 96 + col];
      }
      return 0.f;
    });
  } else {
    fill_tile(tile[w], 96, lane, [&](int row, int col) -> float {
      long e = e0 + (row >> 2); int m = row & 3;
      if (e < E_ && col < 91) {
        long s = src[e];
        return atom[(s * 4 + m) * 91 + col] * rad[e * 96 + col];
      }
      return 0.f;
    });
  }
  __syncthreads();
  for (int nt = 0; nt < 2; ++nt) {
    v8f acc = vzero();
#pragma unroll
    for (int kt = 0; kt < 3; ++kt)
      acc = wmma_bf16(frag_a(tile[w], kt, lane), frag_b(W, kt, nt, 2, lane), acc);
    int col = nt * 16 + (lane & 15), hi = (lane >> 4) << 3;
    if (full) {
#pragma unroll
      for (int r = 0; r < 8; ++r) {
        int row = r + hi; long e = e0 + (row >> 2); int m = row & 3;
        atomicAdd(&a2n[((long)dst[e] * 4 + m) * 32 + col], acc[r]);
      }
    } else {
#pragma unroll
      for (int r = 0; r < 8; ++r) {
        int row = r + hi; long e = e0 + (row >> 2); int m = row & 3;
        if (e < E_) atomicAdd(&a2n[((long)dst[e] * 4 + m) * 32 + col], acc[r]);
      }
    }
  }
}

// ---------------- node update 1 + gated SO3 FFN (wave per node) ----------------
__global__ __launch_bounds__(128) void k_node1(const float* node_emb, const float* a2n,
                                               const float* g_nnu, const float* W1,
                                               const float* Wg, const float* W2,
                                               const float* g_nn, float* node1) {
  __shared__ float nd[4][288], hh[4][576], oo[4][288], gg[4][64];
  int w = threadIdx.x >> 5, lane = threadIdx.x & 31;
  long n = blockIdx.x * 4 + w;
  const float* A = a2n + n * 128;
  float s0 = A[lane] * A[lane], s1 = 0.f;
  for (int i = lane; i < 96; i += 32) { float v = A[32 + i]; s1 += v * v; }
  s0 = wave_sum(s0); s1 = wave_sum(s1);
  float r0 = rsqrtf(s0 / 32.f + 1e-6f), r1 = rsqrtf(s1 / 96.f + 1e-6f);
  const float* NE = node_emb + n * 288;
  for (int i = lane; i < 288; i += 32) {
    int m = i >> 5, c = i & 31;
    float add = 0.f;
    if (m == 0) add = A[c] * r0 * g_nnu[c];
    else if (m < 4) add = A[m * 32 + c] * r1 * g_nnu[32 + c];
    nd[w][i] = NE[i] + add;
  }
  __syncthreads();
  for (int j = lane; j < 64; j += 32) {
    float acc = 0.f;
    for (int c = 0; c < 32; ++c) acc += nd[w][c] * Wg[c * 64 + j];
    gg[w][j] = silu(acc);
  }
  __syncthreads();
  for (int i = lane; i < 576; i += 32) {
    int m = i >> 6, j = i & 63;
    float acc = 0.f;
    for (int c = 0; c < 32; ++c) acc += nd[w][m * 32 + c] * W1[c * 64 + j];
    hh[w][i] = acc * gg[w][j];
  }
  __syncthreads();
  for (int i = lane; i < 288; i += 32) {
    int m = i >> 5, c = i & 31;
    float acc = 0.f;
    for (int j = 0; j < 64; ++j) acc += hh[w][m * 64 + j] * W2[j * 32 + c];
    oo[w][i] = acc;
  }
  __syncthreads();
  float t0 = 0, t1 = 0, t2 = 0;
  for (int i = lane; i < 288; i += 32) {
    float v = oo[w][i]; int m = i >> 5;
    if (m == 0) t0 += v * v; else if (m < 4) t1 += v * v; else t2 += v * v;
  }
  t0 = wave_sum(t0); t1 = wave_sum(t1); t2 = wave_sum(t2);
  float q0 = rsqrtf(t0 / 32.f + 1e-6f), q1 = rsqrtf(t1 / 96.f + 1e-6f), q2 = rsqrtf(t2 / 160.f + 1e-6f);
  for (int i = lane; i < 288; i += 32) {
    int m = i >> 5, c = i & 31;
    float g, q;
    if (m == 0) { g = g_nn[c]; q = q0; }
    else if (m < 4) { g = g_nn[32 + c]; q = q1; }
    else { g = g_nn[64 + c]; q = q2; }
    node1[n * 288 + i] = nd[w][i] + oo[w][i] * q * g;
  }
}

// ---------------- attention logits: silu(ain@Wa1)@Wa2 + segment max ----------------
__global__ __launch_bounds__(256) void k_alpha(const float* node1, const float* ef,
                                               const int* src, const int* dst,
                                               const __bf16* Wa1, const __bf16* Wa2,
                                               float* alpha, float* amax) {
  __shared__ __bf16 ain[8][16 * 192] __attribute__((aligned(32)));
  __shared__ __bf16 mid[8][16 * 32] __attribute__((aligned(32)));
  int w = threadIdx.x >> 5, lane = threadIdx.x & 31;
  long e0 = (long)(blockIdx.x * 8 + w) * 16;
  bool full = (e0 + 16 <= E_);
  if (full) {
    fill_tile(ain[w], 192, lane, [&](int row, int col) -> float {
      long e = e0 + row;
      if (col < 32)      return node1[(long)src[e] * 288 + col];
      else if (col < 64) return node1[(long)dst[e] * 288 + col - 32];
      else               return ef[e * 128 + col - 64];
    });
  } else {
    fill_tile(ain[w], 192, lane, [&](int row, int col) -> float {
      long e = e0 + row;
      if (e >= E_) return 0.f;
      if (col < 32)      return node1[(long)src[e] * 288 + col];
      else if (col < 64) return node1[(long)dst[e] * 288 + col - 32];
      else               return ef[e * 128 + col - 64];
    });
  }
  __syncthreads();
  v8f acc = vzero();
#pragma unroll
  for (int kt = 0; kt < 6; ++kt)
    acc = wmma_bf16(frag_a(ain[w], kt, lane), frag_b(Wa1, kt, 0, 1, lane), acc);
  {
    int col = lane & 15, hi = (lane >> 4) << 3;
#pragma unroll
    for (int r = 0; r < 8; ++r) {
      int row = r + hi;
      tile_store1(mid[w], row, col, silu(acc[r]));
      tile_store1(mid[w], row, col + 16, 0.f);     // zero K-pad
    }
  }
  __syncthreads();
  acc = vzero();
  acc = wmma_bf16(frag_a(mid[w], 0, lane), frag_b(Wa2, 0, 0, 1, lane), acc);
  int col = lane & 15, hi = (lane >> 4) << 3;
  if (col < 8) {
    if (full) {
#pragma unroll
      for (int r = 0; r < 8; ++r) {
        long e = e0 + r + hi;
        alpha[e * 8 + col] = acc[r];
        atomicMaxF(&amax[(long)dst[e] * 8 + col], acc[r]);
      }
    } else {
#pragma unroll
      for (int r = 0; r < 8; ++r) {
        long e = e0 + r + hi;
        if (e < E_) {
          alpha[e * 8 + col] = acc[r];
          atomicMaxF(&amax[(long)dst[e] * 8 + col], acc[r]);
        }
      }
    }
  }
}

__global__ void k_expsum(float* alpha, const float* amax, float* aden, const int* dst) {
  long i = (long)blockIdx.x * blockDim.x + threadIdx.x;
  if (i >= (long)E_ * 8) return;
  long e = i >> 3; int h = i & 7;
  long d = dst[e];
  float ex = __expf(alpha[i] - amax[d * 8 + h]);
  alpha[i] = ex;
  atomicAdd(&aden[d * 8 + h], ex);
}

// ---------------- generic row-batched GEMM: C[R,Nn] = A[R,K] @ B(bf16,frag)[K,Nn] ----------------
__global__ __launch_bounds__(256) void k_rows_gemm(const float* A, const __bf16* B, float* C,
                                                   int R, int K, int Nn) {
  __shared__ __bf16 tile[8][16 * 64] __attribute__((aligned(32)));
  int w = threadIdx.x >> 5, lane = threadIdx.x & 31;
  long r0 = (long)(blockIdx.x * 8 + w) * 16;
  bool full = (r0 + 16 <= R);
  if (full) {
    __builtin_prefetch(A + (r0 + 16) * K, 0, 1);
    fill_tile(tile[w], K, lane, [&](int row, int col) -> float {
      return A[(r0 + row) * K + col];
    });
  } else {
    fill_tile(tile[w], K, lane, [&](int row, int col) -> float {
      long rr = r0 + row;
      return rr < R ? A[rr * K + col] : 0.f;
    });
  }
  __syncthreads();
  int nk = K >> 5, nNT = Nn >> 4;
  for (int nt = 0; nt < nNT; ++nt) {
    v8f acc = vzero();
    for (int kt = 0; kt < nk; ++kt)
      acc = wmma_bf16(frag_a(tile[w], kt, lane), frag_b(B, kt, nt, nNT, lane), acc);
    int col = nt * 16 + (lane & 15), hi = (lane >> 4) << 3;
    if (full) {
#pragma unroll
      for (int r = 0; r < 8; ++r)
        C[(r0 + r + hi) * Nn + col] = acc[r];
    } else {
#pragma unroll
      for (int r = 0; r < 8; ++r) {
        long rr = r0 + r + hi;
        if (rr < R) C[rr * Nn + col] = acc[r];
      }
    }
  }
}

// ---------------- attention scatter: agg[dst] += V[src]*w ----------------
__global__ void k_scatter(const float* alpha, const float* aden, const float* V,
                          const int* src, const int* dst, float* agg) {
  long i = (long)blockIdx.x * blockDim.x + threadIdx.x;
  if (i >= (long)E_ * 72) return;
  long e = i / 72; int rem = (int)(i % 72); int m = rem >> 3; int h = rem & 7;
  long d = dst[e], s = src[e];
  float wgt = alpha[e * 8 + h] * fast_rcp(aden[d * 8 + h] + 1e-9f);
  const float* vp = V + (s * 9 + m) * 64 + h * 8;
  float* ap = agg + (d * 9 + m) * 64 + h * 8;
#pragma unroll
  for (int j = 0; j < 8; ++j) atomicAdd(&ap[j], vp[j] * wgt);
}

// ---------------- node update 2: norm(agg@Wo) + residual -> out ----------------
__global__ __launch_bounds__(128) void k_node2(const float* node1, const float* tmp,
                                               const float* g_attn, float* out_node) {
  int w = threadIdx.x >> 5, lane = threadIdx.x & 31;
  long n = blockIdx.x * 4 + w;
  const float* T = tmp + n * 288;
  float t0 = 0, t1 = 0, t2 = 0;
  for (int i = lane; i < 288; i += 32) {
    float v = T[i]; int m = i >> 5;
    if (m == 0) t0 += v * v; else if (m < 4) t1 += v * v; else t2 += v * v;
  }
  t0 = wave_sum(t0); t1 = wave_sum(t1); t2 = wave_sum(t2);
  float q0 = rsqrtf(t0 / 32.f + 1e-6f), q1 = rsqrtf(t1 / 96.f + 1e-6f), q2 = rsqrtf(t2 / 160.f + 1e-6f);
  for (int i = lane; i < 288; i += 32) {
    int m = i >> 5, c = i & 31;
    float g, q;
    if (m == 0) { g = g_attn[c]; q = q0; }
    else if (m < 4) { g = g_attn[32 + c]; q = q1; }
    else { g = g_attn[64 + c]; q = q2; }
    out_node[n * 288 + i] = node1[n * 288 + i] + T[i] * q * g;
  }
}

// ---------------- edge FFN + LayerNorm + residual ----------------
__global__ __launch_bounds__(128) void k_edge_ffn(const float* node2, const float* ef,
                                                  const int* src, const int* dst,
                                                  const __bf16* W1, const float* b1,
                                                  const __bf16* W2, const float* b2,
                                                  const float* ln_g, const float* ln_b,
                                                  float* out_ef) {
  __shared__ char smem[4 * 12288] __attribute__((aligned(32)));
  int w = threadIdx.x >> 5, lane = threadIdx.x & 31;
  char* base = smem + w * 12288;
  __bf16* ain = (__bf16*)base;            // 6144 B (phase 1)
  float*  ub  = (float*)base;             // 8192 B (phase 2+, overlaps dead ain)
  __bf16* mid = (__bf16*)(base + 8192);   // 4096 B
  long e0 = (long)(blockIdx.x * 4 + w) * 16;
  bool full = (e0 + 16 <= E_);
  if (full) {
    fill_tile(ain, 192, lane, [&](int row, int col) -> float {
      long e = e0 + row;
      if (col < 32)      return node2[(long)src[e] * 288 + col];
      else if (col < 64) return node2[(long)dst[e] * 288 + col - 32];
      else               return ef[e * 128 + col - 64];
    });
  } else {
    fill_tile(ain, 192, lane, [&](int row, int col) -> float {
      long e = e0 + row;
      if (e >= E_) return 0.f;
      if (col < 32)      return node2[(long)src[e] * 288 + col];
      else if (col < 64) return node2[(long)dst[e] * 288 + col - 32];
      else               return ef[e * 128 + col - 64];
    });
  }
  __syncthreads();
  for (int nt = 0; nt < 8; ++nt) {
    v8f acc = vzero();
#pragma unroll
    for (int kt = 0; kt < 6; ++kt)
      acc = wmma_bf16(frag_a(ain, kt, lane), frag_b(W1, kt, nt, 8, lane), acc);
    int col = nt * 16 + (lane & 15), hi = (lane >> 4) << 3;
    float bb = b1[col];
#pragma unroll
    for (int r = 0; r < 8; ++r) {
      float x = acc[r] + bb; if (x < 0.f) x = 0.f;
      tile_store1(mid, r + hi, col, x);
    }
  }
  __syncthreads();
  for (int nt = 0; nt < 8; ++nt) {
    v8f acc = vzero();
#pragma unroll
    for (int kt = 0; kt < 4; ++kt)
      acc = wmma_bf16(frag_a(mid, kt, lane), frag_b(W2, kt, nt, 8, lane), acc);
    int col = nt * 16 + (lane & 15), hi = (lane >> 4) << 3;
    float bb = b2[col];
#pragma unroll
    for (int r = 0; r < 8; ++r) ub[(r + hi) * 128 + col] = acc[r] + bb;
  }
  __syncthreads();
  int row = lane & 15, half = lane >> 4;
  float s = 0.f, s2 = 0.f;
  for (int j = 0; j < 64; ++j) { float v = ub[row * 128 + half * 64 + j]; s += v; s2 += v * v; }
  s += __shfl_xor(s, 16, 32);
  s2 += __shfl_xor(s2, 16, 32);
  float mu = s * (1.f / 128.f);
  float var = s2 * (1.f / 128.f) - mu * mu;
  float rinv = rsqrtf(var + 1e-5f);
  long e = e0 + row;
  if (full) {
    for (int j = 0; j < 64; ++j) {
      int col = half * 64 + j;
      float v = (ub[row * 128 + col] - mu) * rinv * ln_g[col] + ln_b[col];
      out_ef[e * 128 + col] = ef[e * 128 + col] + v;
    }
  } else if (e < E_) {
    for (int j = 0; j < 64; ++j) {
      int col = half * 64 + j;
      float v = (ub[row * 128 + col] - mu) * rinv * ln_g[col] + ln_b[col];
      out_ef[e * 128 + col] = ef[e * 128 + col] + v;
    }
  }
}

// ---------------- node->atom messages: (node2[src]*rad2) @ W_n2a, rows 0..3 only ----------------
__global__ __launch_bounds__(256) void k_n2a(const float* node2, const float* rad2,
                                             const int* src, const int* dst,
                                             const __bf16* W, float* n2a) {
  __shared__ __bf16 tile[8][16 * 32] __attribute__((aligned(32)));
  int w = threadIdx.x >> 5, lane = threadIdx.x & 31;
  long e0 = (long)(blockIdx.x * 8 + w) * 4;
  bool full = (e0 + 4 <= E_);
  if (full) {
    fill_tile(tile[w], 32, lane, [&](int row, int col) -> float {
      long e = e0 + (row >> 2); int m = row & 3;
      return node2[((long)src[e] * 9 + m) * 32 + col] * rad2[e * 32 + col];
    });
  } else {
    fill_tile(tile[w], 32, lane, [&](int row, int col) -> float {
      long e = e0 + (row >> 2); int m = row & 3;
      if (e < E_) return node2[((long)src[e] * 9 + m) * 32 + col] * rad2[e * 32 + col];
      return 0.f;
    });
  }
  __syncthreads();
  for (int nt = 0; nt < 6; ++nt) {
    v8f acc = vzero();
    acc = wmma_bf16(frag_a(tile[w], 0, lane), frag_b(W, 0, nt, 6, lane), acc);
    int col = nt * 16 + (lane & 15), hi = (lane >> 4) << 3;
    if (col < 91) {
      if (full) {
#pragma unroll
        for (int r = 0; r < 8; ++r) {
          int row = r + hi; long e = e0 + (row >> 2); int m = row & 3;
          atomicAdd(&n2a[((long)dst[e] * 4 + m) * 96 + col], acc[r]);
        }
      } else {
#pragma unroll
        for (int r = 0; r < 8; ++r) {
          int row = r + hi; long e = e0 + (row >> 2); int m = row & 3;
          if (e < E_) atomicAdd(&n2a[((long)dst[e] * 4 + m) * 96 + col], acc[r]);
        }
      }
    }
  }
}

// ---------------- atom update + gated SO3 FFN (wave per node) ----------------
__global__ __launch_bounds__(128) void k_atom(const float* atom_emb, const float* n2a,
                                              const float* g_nau, const float* W1,
                                              const float* Wg, const float* W2,
                                              const float* g_na, float* out_atom) {
  __shared__ float at[4][364], hh[4][728], oo[4][364], gg[4][182];
  int w = threadIdx.x >> 5, lane = threadIdx.x & 31;
  long n = blockIdx.x * 4 + w;
  const float* A = n2a + n * 384;
  float s0 = 0.f, s1 = 0.f;
  for (int c = lane; c < 91; c += 32) { float v = A[c]; s0 += v * v; }
  for (int i = lane; i < 273; i += 32) { int m = 1 + i / 91, c = i % 91; float v = A[m * 96 + c]; s1 += v * v; }
  s0 = wave_sum(s0); s1 = wave_sum(s1);
  float r0 = rsqrtf(s0 / 91.f + 1e-6f), r1 = rsqrtf(s1 / 273.f + 1e-6f);
  for (int i = lane; i < 364; i += 32) {
    int m = i / 91, c = i % 91;
    float add = (m == 0) ? A[c] * r0 * g_nau[c] : A[m * 96 + c] * r1 * g_nau[91 + c];
    at[w][i] = atom_emb[n * 364 + i] + add;
  }
  __syncthreads();
  for (int j = lane; j < 182; j += 32) {
    float acc = 0.f;
    for (int c = 0; c < 91; ++c) acc += at[w][c] * Wg[c * 182 + j];
    gg[w][j] = silu(acc);
  }
  __syncthreads();
  for (int i = lane; i < 728; i += 32) {
    int m = i / 182, j = i % 182;
    float acc = 0.f;
    for (int c = 0; c < 91; ++c) acc += at[w][m * 91 + c] * W1[c * 182 + j];
    hh[w][i] = acc * gg[w][j];
  }
  __syncthreads();
  for (int i = lane; i < 364; i += 32) {
    int m = i / 91, c = i % 91;
    float acc = 0.f;
    for (int j = 0; j < 182; ++j) acc += hh[w][m * 182 + j] * W2[j * 91 + c];
    oo[w][i] = acc;
  }
  __syncthreads();
  float t0 = 0.f, t1 = 0.f;
  for (int i = lane; i < 364; i += 32) { float v = oo[w][i]; if (i < 91) t0 += v * v; else t1 += v * v; }
  t0 = wave_sum(t0); t1 = wave_sum(t1);
  float q0 = rsqrtf(t0 / 91.f + 1e-6f), q1 = rsqrtf(t1 / 273.f + 1e-6f);
  for (int i = lane; i < 364; i += 32) {
    int c = i % 91;
    float nrm = (i < 91) ? oo[w][i] * q0 * g_na[c] : oo[w][i] * q1 * g_na[91 + c];
    out_atom[n * 364 + i] = at[w][i] + nrm;
  }
}

// ==================================================================================
extern "C" void kernel_launch(void* const* d_in, const int* in_sizes, int n_in,
                              void* d_out, int out_size, void* d_ws, size_t ws_size,
                              hipStream_t stream) {
  const float* atom_emb = (const float*)d_in[0];
  const float* node_emb = (const float*)d_in[1];
  const float* edge_feat = (const float*)d_in[2];
  const int*   ei = (const int*)d_in[3];
  const int* src = ei, * dst = ei + E_;

  const float* W_rad_a2n = (const float*)d_in[4];
  const float* b_rad_a2n = (const float*)d_in[5];
  const float* W_a2n = (const float*)d_in[6];
  const float* g_nnu = (const float*)d_in[7];
  const float* W1_nf = (const float*)d_in[8];
  const float* Wg_nf = (const float*)d_in[9];
  const float* W2_nf = (const float*)d_in[10];
  const float* g_nn = (const float*)d_in[11];
  const float* Wa1 = (const float*)d_in[12];
  const float* Wa2 = (const float*)d_in[13];
  const float* Wv = (const float*)d_in[14];
  const float* Wo = (const float*)d_in[15];
  const float* g_attn = (const float*)d_in[16];
  const float* Wff1 = (const float*)d_in[17];
  const float* bff1 = (const float*)d_in[18];
  const float* Wff2 = (const float*)d_in[19];
  const float* bff2 = (const float*)d_in[20];
  const float* ln_g = (const float*)d_in[21];
  const float* ln_b = (const float*)d_in[22];
  const float* W_rad_n2a = (const float*)d_in[23];
  const float* b_rad_n2a = (const float*)d_in[24];
  const float* W_n2a = (const float*)d_in[25];
  const float* g_nau = (const float*)d_in[26];
  const float* W1_af = (const float*)d_in[27];
  const float* Wg_af = (const float*)d_in[28];
  const float* W2_af = (const float*)d_in[29];
  const float* g_na = (const float*)d_in[30];

  float* out_atom = (float*)d_out;                  // N*4*91
  float* out_node = out_atom + (size_t)N_ * 4 * 91; // N*9*32
  float* out_ef = out_node + (size_t)N_ * 9 * 32;   // E*128

  // ---- workspace carve (256B aligned), with aliasing ----
  char* base = (char*)d_ws;
  size_t off = 0;
  auto alloc = [&](size_t bytes) -> char* {
    char* r = base + off; off += (bytes + 255) & ~(size_t)255; return r;
  };
  __bf16* WR1b  = (__bf16*)alloc(128 * 96 * 2);
  __bf16* WA2Nb = (__bf16*)alloc(96 * 32 * 2);
  __bf16* WA1b  = (__bf16*)alloc(192 * 16 * 2);
  __bf16* WA2b  = (__bf16*)alloc(32 * 16 * 2);
  __bf16* WVb   = (__bf16*)alloc(32 * 64 * 2);
  __bf16* WOb   = (__bf16*)alloc(64 * 32 * 2);
  __bf16* WFF1b = (__bf16*)alloc(192 * 128 * 2);
  __bf16* WFF2b = (__bf16*)alloc(128 * 128 * 2);
  __bf16* WR2b  = (__bf16*)alloc(128 * 32 * 2);
  __bf16* WN2Ab = (__bf16*)alloc(32 * 96 * 2);
  float* bias96 = (float*)alloc(96 * 4);
  float* rad   = (float*)alloc((size_t)E_ * 96 * 4);   // alias: V after k_a2n done
  float* Vbuf  = rad;                                  // N*9*64 = 57.6MB <= 76.8MB
  float* a2n   = (float*)alloc((size_t)N_ * 128 * 4);
  float* node1 = (float*)alloc((size_t)N_ * 288 * 4);
  float* alpha = (float*)alloc((size_t)E_ * 8 * 4);
  float* amax  = (float*)alloc((size_t)N_ * 8 * 4);
  float* aden  = (float*)alloc((size_t)N_ * 8 * 4);
  float* agg   = (float*)alloc((size_t)N_ * 576 * 4);  // alias: n2a after k_Wo done
  float* n2a   = agg;                                  // N*4*96 = 38.4MB <= 57.6MB
  float* tmpb  = (float*)alloc((size_t)N_ * 288 * 4);  // alias: rad2 after k_node2 done
  float* rad2  = tmpb;                                 // E*32 = 25.6MB <= 28.8MB

  auto cvt = [&](const float* s, __bf16* d, int K, int Nsrc, int Kp, int Np) {
    int tot = Kp * Np;
    k_cvt<<<(tot + 255) / 256, 256, 0, stream>>>(s, d, K, Nsrc, Kp, Np);
  };
  cvt(W_rad_a2n, WR1b, 128, 91, 128, 96);
  cvt(W_a2n,     WA2Nb, 91, 32, 96, 32);
  cvt(Wa1,       WA1b, 192, 16, 192, 16);
  cvt(Wa2,       WA2b, 16, 8, 32, 16);
  cvt(Wv,        WVb, 32, 64, 32, 64);
  cvt(Wo,        WOb, 64, 32, 64, 32);
  cvt(Wff1,      WFF1b, 192, 128, 192, 128);
  cvt(Wff2,      WFF2b, 128, 128, 128, 128);
  cvt(W_rad_n2a, WR2b, 128, 32, 128, 32);
  cvt(W_n2a,     WN2Ab, 32, 91, 32, 96);
  k_padvec<<<1, 96, 0, stream>>>(b_rad_a2n, bias96, 91, 96);

  hipMemsetAsync(a2n, 0, (size_t)N_ * 128 * 4, stream);
  hipMemsetAsync(aden, 0, (size_t)N_ * 8 * 4, stream);
  k_fill<<<(N_ * 8 + 255) / 256, 256, 0, stream>>>(amax, -3.402823466e38f, (long)N_ * 8);

  // 1) rad = silu(ef @ W_rad_a2n + b)   [E x 96 padded]
  k_edge_lin<<<(12500 + 7) / 8, 256, 0, stream>>>(edge_feat, WR1b, bias96, rad, 96, 96);
  // 2) a2n scatter (4 edges/wave)
  k_a2n<<<50000 / 8, 256, 0, stream>>>(atom_emb, rad, src, dst, WA2Nb, a2n);
  // 3) node update 1 + node FFN
  k_node1<<<N_ / 4, 128, 0, stream>>>(node_emb, a2n, g_nnu, W1_nf, Wg_nf, W2_nf, g_nn, node1);
  // 4) attention logits + segment max
  k_alpha<<<(12500 + 7) / 8, 256, 0, stream>>>(node1, edge_feat, src, dst, WA1b, WA2b, alpha, amax);
  // 5) exp + segment sum
  k_expsum<<<(E_ * 8) / 256, 256, 0, stream>>>(alpha, amax, aden, dst);
  // 6) V = node1_flat @ Wv   [225000 x 64]
  {
    int R = N_ * 9, waves = (R + 15) / 16;
    k_rows_gemm<<<(waves + 7) / 8, 256, 0, stream>>>(node1, WVb, Vbuf, R, 32, 64);
  }
  // 7) weighted scatter
  hipMemsetAsync(agg, 0, (size_t)N_ * 576 * 4, stream);
  k_scatter<<<((long)E_ * 72) / 256, 256, 0, stream>>>(alpha, aden, Vbuf, src, dst, agg);
  // 8) tmp = agg_flat @ Wo
  {
    int R = N_ * 9, waves = (R + 15) / 16;
    k_rows_gemm<<<(waves + 7) / 8, 256, 0, stream>>>(agg, WOb, tmpb, R, 64, 32);
  }
  // 9) node update 2 -> out_node (final node_emb)
  k_node2<<<N_ / 4, 128, 0, stream>>>(node1, tmpb, g_attn, out_node);
  // 10) edge FFN + layernorm -> out_ef (final edge_feat)
  k_edge_ffn<<<12500 / 4, 128, 0, stream>>>(out_node, edge_feat, src, dst,
                                            WFF1b, bff1, WFF2b, bff2, ln_g, ln_b, out_ef);
  // 11) rad2 = silu(out_ef @ W_rad_n2a + b)   [E x 32]
  k_edge_lin<<<(12500 + 7) / 8, 256, 0, stream>>>(out_ef, WR2b, b_rad_n2a, rad2, 32, 32);
  // 12) n2a scatter (rows 0..3 only)
  hipMemsetAsync(n2a, 0, (size_t)N_ * 384 * 4, stream);
  k_n2a<<<50000 / 8, 256, 0, stream>>>(out_node, rad2, src, dst, WN2Ab, n2a);
  // 13) atom update + atom FFN -> out_atom (final atom_emb)
  k_atom<<<N_ / 4, 128, 0, stream>>>(atom_emb, n2a, g_nau, W1_af, Wg_af, W2_af, g_na, out_atom);
}
